// LiquidCell_16432544875337
// MI455X (gfx1250) — matrix operand order repaired and checked by reading, not compiled, not run on hardware
//
#include <hip/hip_runtime.h>

#define BATCH   32768
#define SDIM    512
#define IDIM    256
#define DDIM    768     // SDIM + IDIM
#define HDIM    1024
#define DT_C    0.05f
#define TAUMIN  0.01f
#define LNEPS   1e-5f

// WMMA tile config
#define MT 128
#define NT 128
#define KT 32
#define LDSS 40        // padded row stride in bf16 elements (80B, bank-spreading)

typedef __attribute__((ext_vector_type(16))) __bf16 v16bf;
typedef __attribute__((ext_vector_type(8)))  float  v8f;

union FragBF { uint4 u4[2]; v16bf v; };

__device__ __forceinline__ unsigned short f2bf(float f) {
    union { float f; unsigned u; } v; v.f = f;
    unsigned r = v.u + 0x7FFFu + ((v.u >> 16) & 1u);   // RNE
    return (unsigned short)(r >> 16);
}

// CDNA5 async copy: global -> LDS, 16B per lane, tracked by ASYNCcnt.
// lds_off = wave-relative LDS byte address (low 32 bits of generic shared ptr).
__device__ __forceinline__ void async_b128(unsigned lds_off, const void* gaddr) {
    asm volatile("global_load_async_to_lds_b128 %0, %1, off"
                 :: "v"(lds_off), "v"(gaddr) : "memory");
}
__device__ __forceinline__ void wait_async0() {
    asm volatile("s_wait_asynccnt 0x0" ::: "memory");
}

// ---------------------------------------------------------------------------
// Weight transpose + fp32 -> bf16:  in [K][N] row-major  ->  out [N][K]
// ---------------------------------------------------------------------------
__global__ void transpose_to_bf16(const float* __restrict__ in,
                                  unsigned short* __restrict__ out,
                                  int K, int N) {
    size_t idx = (size_t)blockIdx.x * blockDim.x + threadIdx.x;
    size_t total = (size_t)K * N;
    if (idx >= total) return;
    int n = (int)(idx / K);
    int k = (int)(idx % K);
    out[idx] = f2bf(in[(size_t)k * N + n]);
}

__global__ void concat_bias(const float* __restrict__ tb,
                            const float* __restrict__ gb,
                            float* __restrict__ btg) {
    int j = blockIdx.x * blockDim.x + threadIdx.x;
    if (j < 2 * SDIM) btg[j] = (j < SDIM) ? tb[j] : gb[j - SDIM];
}

// xu[:, :512] = bf16(x); xu[:, 512:768] = bf16(u)
__global__ void pack_xu(const float* __restrict__ x, const float* __restrict__ u,
                        unsigned short* __restrict__ xu) {
    size_t idx = (size_t)blockIdx.x * blockDim.x + threadIdx.x;   // B*DDIM threads
    int m = (int)(idx / DDIM);
    int j = (int)(idx % DDIM);
    float v = (j < SDIM) ? x[(size_t)m * SDIM + j] : u[(size_t)m * IDIM + (j - SDIM)];
    xu[idx] = f2bf(v);
}

// ---------------------------------------------------------------------------
// Fused bf16 WMMA GEMM with async double-buffered LDS staging.
//   acc = A[M x K](bf16) @ WT[N x K]^T(bf16) + bias
// MODE 0: out_bf  = bf16(silu(acc))                     (h1 = silu(xu@fW1+fb1))
// MODE 1: tg[:, :512] = max(softplus,TAU_MIN); [:,512:] = sigmoid
// MODE 2: step-1 combine: f=tanh(acc)*gate; dx1=DT*(f-x)/tau; x1=x+dx1
//         store dx1, x1 (f32) and refresh xu[:, :512] = bf16(x1)
// MODE 3: step-2 combine: dx2=DT*(f-x1)/tau2; y = x + 0.5*(dx1+dx2)
// ---------------------------------------------------------------------------
template <int MODE>
__global__ __launch_bounds__(256)
void gemm_bf16(const unsigned short* __restrict__ A, int K,
               const unsigned short* __restrict__ WT,
               const float* __restrict__ bias, int N,
               unsigned short* __restrict__ out_bf,
               float* __restrict__ tg,
               const float* __restrict__ xin,
               float* __restrict__ dx1,
               float* __restrict__ x1buf,
               unsigned short* __restrict__ xu_out,
               float* __restrict__ yout) {
    __shared__ __align__(16) unsigned short Asm[2][MT * LDSS];
    __shared__ __align__(16) unsigned short Bsm[2][NT * LDSS];

    const int tid  = threadIdx.x;
    const int lane = tid & 31;
    const int wid  = tid >> 5;
    const int half = lane >> 4;     // 0: lanes 0-15, 1: lanes 16-31
    const int l16  = lane & 15;

    const int m0 = blockIdx.y * MT;
    const int n0 = blockIdx.x * NT;
    const int wm = (wid & 3) * 32;  // wave's row offset inside tile  (4 waves in M)
    const int wn = (wid >> 2) * 64; // wave's col offset inside tile  (2 waves in N)

    // Per-thread copy assignment: 2 x 16B chunks for A and B tiles each.
    const int r0 = (tid) >> 2, s0 = (tid) & 3;
    const int r1 = (tid + 256) >> 2, s1 = (tid + 256) & 3;

    const unsigned baseA0 = (unsigned)(uintptr_t)(&Asm[0][0]);
    const unsigned baseA1 = (unsigned)(uintptr_t)(&Asm[1][0]);
    const unsigned baseB0 = (unsigned)(uintptr_t)(&Bsm[0][0]);
    const unsigned baseB1 = (unsigned)(uintptr_t)(&Bsm[1][0]);

    v8f zero = {};
    v8f acc[2][4];
#pragma unroll
    for (int i = 0; i < 2; ++i)
#pragma unroll
        for (int j = 0; j < 4; ++j) acc[i][j] = zero;

    const int nsteps = K / KT;

    // Prologue: stage tile 0 into buffer 0.
    async_b128(baseA0 + (unsigned)((r0 * LDSS + s0 * 8) * 2),
               A + (size_t)(m0 + r0) * K + s0 * 8);
    async_b128(baseA0 + (unsigned)((r1 * LDSS + s1 * 8) * 2),
               A + (size_t)(m0 + r1) * K + s1 * 8);
    async_b128(baseB0 + (unsigned)((r0 * LDSS + s0 * 8) * 2),
               WT + (size_t)(n0 + r0) * K + s0 * 8);
    async_b128(baseB0 + (unsigned)((r1 * LDSS + s1 * 8) * 2),
               WT + (size_t)(n0 + r1) * K + s1 * 8);
    wait_async0();
    __syncthreads();

    for (int ks = 0; ks < nsteps; ++ks) {
        const int cur = ks & 1;
        // Kick off async copies for the next K-tile into the other buffer.
        if (ks + 1 < nsteps) {
            const int k1 = (ks + 1) * KT;
            const unsigned bA = cur ? baseA0 : baseA1;
            const unsigned bB = cur ? baseB0 : baseB1;
            async_b128(bA + (unsigned)((r0 * LDSS + s0 * 8) * 2),
                       A + (size_t)(m0 + r0) * K + k1 + s0 * 8);
            async_b128(bA + (unsigned)((r1 * LDSS + s1 * 8) * 2),
                       A + (size_t)(m0 + r1) * K + k1 + s1 * 8);
            async_b128(bB + (unsigned)((r0 * LDSS + s0 * 8) * 2),
                       WT + (size_t)(n0 + r0) * K + k1 + s0 * 8);
            async_b128(bB + (unsigned)((r1 * LDSS + s1 * 8) * 2),
                       WT + (size_t)(n0 + r1) * K + k1 + s1 * 8);
        }

        // Compute on the current buffer (copies in flight target the other one).
#pragma unroll
        for (int mi = 0; mi < 2; ++mi) {
            // A fragment (16x32 bf16): lanes 0-15 take K 0-7 & 16-23, lanes 16-31 take 8-15 & 24-31
            FragBF af;
            const unsigned short* abase = &Asm[cur][(wm + mi * 16 + l16) * LDSS];
            af.u4[0] = *(const uint4*)(abase + half * 8);
            af.u4[1] = *(const uint4*)(abase + 16 + half * 8);
#pragma unroll
            for (int ni = 0; ni < 4; ++ni) {
                // B fragment (32x16): lanes 0-15 take K 0-7 & 16-23, lanes 16-31 take 8-15 & 24-31, n = lane%16
                FragBF bfr;
                const unsigned short* bbase = &Bsm[cur][(wn + ni * 16 + l16) * LDSS];
                bfr.u4[0] = *(const uint4*)(bbase + half * 8);
                bfr.u4[1] = *(const uint4*)(bbase + 16 + half * 8);
                acc[mi][ni] = __builtin_amdgcn_wmma_f32_16x16x32_bf16(
                    false, af.v, false, bfr.v, (short)0, acc[mi][ni], false, false);
            }
        }

        // Next tile's copies done (mine) + everyone else's (barrier).
        wait_async0();
        __syncthreads();
    }

    // Epilogue. C/D layout: VGPR r -> row = r + 8*half, col = lane%16.
#pragma unroll
    for (int mi = 0; mi < 2; ++mi) {
#pragma unroll
        for (int ni = 0; ni < 4; ++ni) {
#pragma unroll
            for (int r = 0; r < 8; ++r) {
                int row = m0 + wm + mi * 16 + r + 8 * half;
                int col = n0 + wn + ni * 16 + l16;
                float v = acc[mi][ni][r] + bias[col];
                if (MODE == 0) {
                    float s = v / (1.f + __expf(-v));               // silu
                    out_bf[(size_t)row * N + col] = f2bf(s);
                } else if (MODE == 1) {
                    float o;
                    if (col < SDIM) {
                        float sp = (v > 20.f) ? v : log1pf(__expf(v));  // softplus
                        o = fmaxf(sp, TAUMIN);
                    } else {
                        o = 1.f / (1.f + __expf(-v));                   // sigmoid
                    }
                    tg[(size_t)row * (2 * SDIM) + col] = o;
                } else {
                    float tau  = tg[(size_t)row * (2 * SDIM) + col];
                    float gate = tg[(size_t)row * (2 * SDIM) + SDIM + col];
                    float f = tanhf(v) * gate;
                    if (MODE == 2) {
                        float xv  = xin[(size_t)row * SDIM + col];
                        float dx  = DT_C * (f - xv) / tau;
                        float x1v = xv + dx;
                        dx1[(size_t)row * SDIM + col]   = dx;
                        x1buf[(size_t)row * SDIM + col] = x1v;
                        xu_out[(size_t)row * DDIM + col] = f2bf(x1v);
                    } else {
                        float x1v = x1buf[(size_t)row * SDIM + col];
                        float dx2 = DT_C * (f - x1v) / tau;
                        yout[(size_t)row * SDIM + col] =
                            xin[(size_t)row * SDIM + col] +
                            0.5f * (dx1[(size_t)row * SDIM + col] + dx2);
                    }
                }
            }
        }
    }
}

// ---------------------------------------------------------------------------
// LayerNorm over last dim (512). One wave32 per row, 8 rows per 256-thr block.
// ---------------------------------------------------------------------------
__global__ __launch_bounds__(256)
void layernorm_rows(const float* __restrict__ y, const float* __restrict__ g,
                    const float* __restrict__ b, float* __restrict__ out) {
    int wid  = threadIdx.x >> 5;
    int lane = threadIdx.x & 31;
    int row  = blockIdx.x * 8 + wid;
    const float* yr = y + (size_t)row * SDIM;
    float vals[16];
    float s = 0.f;
#pragma unroll
    for (int i = 0; i < 16; ++i) { vals[i] = yr[lane + i * 32]; s += vals[i]; }
#pragma unroll
    for (int off = 16; off; off >>= 1) s += __shfl_xor(s, off, 32);
    float mu = s * (1.f / SDIM);
    float vs = 0.f;
#pragma unroll
    for (int i = 0; i < 16; ++i) { float d = vals[i] - mu; vs += d * d; }
#pragma unroll
    for (int off = 16; off; off >>= 1) vs += __shfl_xor(vs, off, 32);
    float inv = rsqrtf(vs * (1.f / SDIM) + LNEPS);
    float* orow = out + (size_t)row * SDIM;
#pragma unroll
    for (int i = 0; i < 16; ++i) {
        int c = lane + i * 32;
        orow[c] = (vals[i] - mu) * inv * g[c] + b[c];
    }
}

// ---------------------------------------------------------------------------
extern "C" void kernel_launch(void* const* d_in, const int* in_sizes, int n_in,
                              void* d_out, int out_size, void* d_ws, size_t ws_size,
                              hipStream_t stream) {
    const float* x    = (const float*)d_in[0];
    const float* u    = (const float*)d_in[1];
    const float* fW1  = (const float*)d_in[2];
    const float* fb1  = (const float*)d_in[3];
    const float* fW2  = (const float*)d_in[4];
    const float* fb2  = (const float*)d_in[5];
    const float* tW   = (const float*)d_in[6];
    const float* tb   = (const float*)d_in[7];
    const float* gW   = (const float*)d_in[8];
    const float* gb   = (const float*)d_in[9];
    const float* ln_g = (const float*)d_in[10];
    const float* ln_b = (const float*)d_in[11];
    float* out = (float*)d_out;

    char*  ws  = (char*)d_ws;
    size_t off = 0;
    auto alloc = [&](size_t bytes) -> void* {
        void* p = ws + off;
        off = (off + bytes + 255) & ~(size_t)255;
        return p;
    };
    unsigned short* W1T  = (unsigned short*)alloc((size_t)HDIM * DDIM * 2);  // fW1^T  [1024][768]
    unsigned short* W2T  = (unsigned short*)alloc((size_t)SDIM * HDIM * 2);  // fW2^T  [512][1024]
    unsigned short* WTGT = (unsigned short*)alloc((size_t)2 * SDIM * DDIM * 2); // [tW|gW]^T [1024][768]
    float*          btg  = (float*)alloc((size_t)2 * SDIM * 4);
    unsigned short* xu   = (unsigned short*)alloc((size_t)BATCH * DDIM * 2);
    unsigned short* h1   = (unsigned short*)alloc((size_t)BATCH * HDIM * 2);
    float*          tg   = (float*)alloc((size_t)BATCH * 2 * SDIM * 4);
    float*          dx1  = (float*)alloc((size_t)BATCH * SDIM * 4);
    float*          x1   = (float*)alloc((size_t)BATCH * SDIM * 4);
    float*          yb   = (float*)alloc((size_t)BATCH * SDIM * 4);
    (void)ws_size; (void)in_sizes; (void)n_in; (void)out_size;

    // Weight prep (small, L2-resident afterwards)
    transpose_to_bf16<<<(HDIM * DDIM + 255) / 256, 256, 0, stream>>>(fW1, W1T, DDIM, HDIM);
    transpose_to_bf16<<<(SDIM * HDIM + 255) / 256, 256, 0, stream>>>(fW2, W2T, HDIM, SDIM);
    transpose_to_bf16<<<(SDIM * DDIM + 255) / 256, 256, 0, stream>>>(tW, WTGT, DDIM, SDIM);
    transpose_to_bf16<<<(SDIM * DDIM + 255) / 256, 256, 0, stream>>>(gW, WTGT + (size_t)SDIM * DDIM, DDIM, SDIM);
    concat_bias<<<4, 256, 0, stream>>>(tb, gb, btg);
    pack_xu<<<(BATCH * DDIM) / 256, 256, 0, stream>>>(x, u, xu);

    dim3 gH(HDIM / NT, BATCH / MT);   // (8, 256)
    dim3 gS(SDIM / NT, BATCH / MT);   // (4, 256)

    // ---- Heun step 1 ----
    gemm_bf16<0><<<gH, 256, 0, stream>>>(xu, DDIM, W1T, fb1, HDIM, h1, nullptr, nullptr, nullptr, nullptr, nullptr, nullptr);
    gemm_bf16<1><<<gH, 256, 0, stream>>>(xu, DDIM, WTGT, btg, 2 * SDIM, nullptr, tg, nullptr, nullptr, nullptr, nullptr, nullptr);
    gemm_bf16<2><<<gS, 256, 0, stream>>>(h1, HDIM, W2T, fb2, SDIM, nullptr, tg, x, dx1, x1, xu, nullptr);
    // ---- Heun step 2 ----
    gemm_bf16<0><<<gH, 256, 0, stream>>>(xu, DDIM, W1T, fb1, HDIM, h1, nullptr, nullptr, nullptr, nullptr, nullptr, nullptr);
    gemm_bf16<1><<<gH, 256, 0, stream>>>(xu, DDIM, WTGT, btg, 2 * SDIM, nullptr, tg, nullptr, nullptr, nullptr, nullptr, nullptr);
    gemm_bf16<3><<<gS, 256, 0, stream>>>(h1, HDIM, W2T, fb2, SDIM, nullptr, tg, x, dx1, x1, nullptr, yb);

    layernorm_rows<<<BATCH / 8, 256, 0, stream>>>(yb, ln_g, ln_b, out);
}
